// Grid_fun_4303557230990
// MI455X (gfx1250) — matrix operand order repaired and checked by reading, not compiled, not run on hardware
//
#include <hip/hip_runtime.h>
#include <hip/hip_bf16.h>

typedef _Float16 v16h __attribute__((ext_vector_type(16)));
typedef float    v8f  __attribute__((ext_vector_type(8)));

#define WAVES_PER_BLOCK 8
#define BLOCK_SIZE (WAVES_PER_BLOCK * 32)

struct F3 { float x, y, z; };  // 12-byte contiguous load/store unit

// out[N,3] = feat[N,16] @ a_eff[16,3], feat = outer([x,1],[x,1]).reshape(16),
// a_eff[k] = sum_{r<6} a[16r+k] (the 6x tiling in the reference collapses).
// Computed transposed: D[chan][point] = a_eff^T (A, 16x32) x feat^T (B, 32x16)
// so each lo-half lane owns one point: builds all 16 features into its B
// column and gets channels 0..2 back in acc[0..2] (consecutive VGPRs) for a
// single coalesced 12B store. One v_wmma_f32_16x16x32_f16 per 16-point tile.
__global__ void grid_fun_wmma_kernel(const float* __restrict__ x,
                                     const float* __restrict__ a,
                                     float* __restrict__ out,
                                     int n, int numTiles) {
    __shared__ float s_aeff[48];  // a_eff[k][c], k<16, c<3

    // Block-cooperative reduction a[96,3] -> a_eff[16,3] (L2-cached loads).
    int t = threadIdx.x;
    if (t < 48) {
        int k = t / 3, c = t % 3;
        float s = 0.f;
#pragma unroll
        for (int r = 0; r < 6; ++r) s += a[(16 * r + k) * 3 + c];
        s_aeff[t] = s;
    }
    __syncthreads();

    const int lane = threadIdx.x & 31;
    const int hi   = lane >> 4;   // 0: lanes 0-15, 1: lanes 16-31
    const int m    = lane & 15;   // A row (channel) == B/D column (point)

    // A fragment (16x32 f16), loop-invariant: A[c][k] = a_eff[k][c] for c<3,
    // k<16; rows 3..15 and K>=16 are zero. Contiguous K order (matches B):
    // lo lanes hold K=0..15 in halves 0..15; hi lanes hold K=16..31 (zero).
    v16h afrag;
#pragma unroll
    for (int e = 0; e < 16; ++e) afrag[e] = (_Float16)0.f;
    if (m < 3 && hi == 0) {
#pragma unroll
        for (int e = 0; e < 16; ++e) {
            int k = e;                          // K < 16 region only
            afrag[e] = (_Float16)s_aeff[k * 3 + m];
        }
    }

    const int waveGlobal = blockIdx.x * WAVES_PER_BLOCK + (threadIdx.x >> 5);
    const int waveStride = gridDim.x * WAVES_PER_BLOCK;

    for (int tile = waveGlobal; tile < numTiles; tile += waveStride) {
        const int  base = tile * 16;
        const int  p    = base + m;             // lo-half lane's point
        const bool act  = (hi == 0) && (p < n);

        float z[4] = {0.f, 0.f, 0.f, 0.f};
        if (act) {
            F3 v = *(const F3*)(x + 3 * p);     // 12B contiguous, coalesced
            z[0] = v.x; z[1] = v.y; z[2] = v.z; z[3] = 1.f;
        }

        // B fragment (32x16 f16): column = m. Lo lanes hold K=0..15 packed in
        // halves 0..15 = all 16 features of this point; hi lanes (K=16..31)
        // are the zero pad (z == 0 there, so products vanish).
        v16h bfrag;
#pragma unroll
        for (int e = 0; e < 16; ++e)
            bfrag[e] = (_Float16)(z[e >> 2] * z[e & 3]);

        v8f acc = {};
        // (neg_a, A, neg_b, B, c_mod, C, reuse_a, reuse_b); EXEC all-1s here.
        acc = __builtin_amdgcn_wmma_f32_16x16x32_f16(
            false, afrag, false, bfrag, (short)0, acc, false, false);

        // D: lo lane m holds D[0..2][m] = out channels of point p in
        // acc[0..2] (consecutive VGPRs) -> single contiguous 12B store.
        if (act) {
            F3 o; o.x = acc[0]; o.y = acc[1]; o.z = acc[2];
            *(F3*)(out + 3 * p) = o;
        }
    }
}

extern "C" void kernel_launch(void* const* d_in, const int* in_sizes, int n_in,
                              void* d_out, int out_size, void* d_ws, size_t ws_size,
                              hipStream_t stream) {
    const float* x   = (const float*)d_in[0];   // [N,3] float32
    const float* a   = (const float*)d_in[1];   // [96,3] float32
    float*       out = (float*)d_out;           // [N,3] float32

    int n        = in_sizes[0] / 3;
    int numTiles = (n + 15) / 16;
    int blocks   = (numTiles + WAVES_PER_BLOCK - 1) / WAVES_PER_BLOCK;
    if (blocks < 1) blocks = 1;

    grid_fun_wmma_kernel<<<blocks, BLOCK_SIZE, 0, stream>>>(x, a, out, n, numTiles);
}
